// GroupedQueryAttention_84473416778119
// MI455X (gfx1250) — compile-verified, matrix operand order checked by reading, not run on hardware
//
#include <hip/hip_runtime.h>

// Problem constants (match reference)
#define BQ_B   2
#define BQ_N   2048
#define D_IN   2048
#define D_OUT  2048
#define NH     32
#define NKV    8
#define NG     4
#define DHD    64

typedef __attribute__((ext_vector_type(16))) __bf16          v16bf;
typedef __attribute__((ext_vector_type(8)))  float           v8f;
typedef __attribute__((ext_vector_type(8)))  unsigned short  v8us;
typedef __attribute__((ext_vector_type(16))) unsigned short  v16us;
typedef __attribute__((ext_vector_type(4)))  unsigned short  v4us;

__device__ __forceinline__ unsigned short f2bf(float f) {
    unsigned int u = __float_as_uint(f);
    u += 0x7FFFu + ((u >> 16) & 1u);         // round-to-nearest-even
    return (unsigned short)(u >> 16);
}

// Load one 16x32 bf16 fragment (A layout; B uses the same layout on an [n][k]
// tile). Lane l<16 holds row/col l with K {kb..kb+7, kb+16..kb+23}; lane
// l>=16 holds K {kb+8..kb+15, kb+24..kb+31}.  Two ds_load_b128 per lane.
__device__ __forceinline__ v16bf frag_ld(const unsigned short* base, int row,
                                         int ldk, int kbase, int half) {
    const v8us lo = *(const v8us*)(base + row * ldk + kbase + half * 8);
    const v8us hi = *(const v8us*)(base + row * ldk + kbase + 16 + half * 8);
    v16us c = __builtin_shufflevector(lo, hi, 0,1,2,3,4,5,6,7,8,9,10,11,12,13,14,15);
    return __builtin_bit_cast(v16bf, c);
}

__device__ __forceinline__ v8f wmma_bf16(v16bf a, v16bf b, v8f c) {
    return __builtin_amdgcn_wmma_f32_16x16x32_bf16(false, a, false, b,
                                                   (short)0, c, false, false);
}

// Async global -> LDS copy (16 bytes / lane), tracked by ASYNCcnt.
__device__ __forceinline__ void async_cp_b128(unsigned lds_addr, unsigned goff,
                                              const void* sbase) {
    asm volatile("global_load_async_to_lds_b128 %0, %1, %2 offset:0"
                 :: "v"(lds_addr), "v"(goff), "s"(sbase) : "memory");
}
__device__ __forceinline__ void wait_async(int pending) {
    if (pending) asm volatile("s_wait_asynccnt 4" ::: "memory");
    else         asm volatile("s_wait_asynccnt 0" ::: "memory");
}

// ---------------------------------------------------------------------------
// Pre-pass 1: fp32 -> bf16 elementwise (n multiple of 1024)
// ---------------------------------------------------------------------------
__global__ void __launch_bounds__(256)
cvt_f32_bf16(const float* __restrict__ in, unsigned short* __restrict__ out) {
    size_t i = ((size_t)blockIdx.x * 256 + threadIdx.x) * 4;
    float4 d = *(const float4*)(in + i);
    v4us o = { f2bf(d.x), f2bf(d.y), f2bf(d.z), f2bf(d.w) };
    *(v4us*)(out + i) = o;
}

// ---------------------------------------------------------------------------
// Pre-pass 2: W[K][N] fp32 -> Wt[N][K] bf16 (64x64 LDS-tiled transpose)
// ---------------------------------------------------------------------------
__global__ void __launch_bounds__(256)
transpose_f32_bf16(const float* __restrict__ W, unsigned short* __restrict__ Wt,
                   int K, int N) {
    __shared__ unsigned short tile[64][65];
    const int tid = threadIdx.x;
    const int kb = blockIdx.y * 64, nb = blockIdx.x * 64;
#pragma unroll
    for (int i = 0; i < 16; ++i) {
        int idx = tid + i * 256;
        int r = idx >> 6, c = idx & 63;
        tile[r][c] = f2bf(W[(size_t)(kb + r) * N + nb + c]);
    }
    __syncthreads();
#pragma unroll
    for (int i = 0; i < 16; ++i) {
        int idx = tid + i * 256;
        int r = idx >> 6, c = idx & 63;
        Wt[(size_t)(nb + r) * K + kb + c] = tile[c][r];
    }
}

// ---------------------------------------------------------------------------
// GEMM: C[M,N] = A[M,K] @ Bt[N,K]^T (+bias). A,Bt bf16 in global; double-
// buffered LDS staging via global_load_async_to_lds_b128 (ASYNCcnt).
// 128x128 block, 256 threads (8 waves), each wave 64x32 (4x2 WMMA tiles).
// ---------------------------------------------------------------------------
template<bool OUT_BF16>
__global__ void __launch_bounds__(256)
gemm_wmma_async(const unsigned short* __restrict__ Abf,
                const unsigned short* __restrict__ Bt,
                const float* __restrict__ bias, void* __restrict__ Cout,
                int M, int Nn, int K)
{
    __shared__ __align__(16) unsigned short lds_a[2][128 * 32]; // [m][k] bf16
    __shared__ __align__(16) unsigned short lds_b[2][128 * 32]; // [n][k] bf16

    const int tid  = threadIdx.x;
    const int lane = tid & 31;
    const int wave = tid >> 5;
    const int half = lane >> 4;
    const int lrow = lane & 15;
    const int wm   = (wave >> 2) * 64;
    const int wn   = (wave & 3) * 32;
    const int m0   = blockIdx.y * 128;
    const int n0   = blockIdx.x * 128;

    // Per-thread copy slots: two 16B chunks of each 8KB tile.
    const unsigned off0 = tid * 16u, off1 = off0 + 4096u;
    const unsigned r0 = off0 >> 6, kb0 = off0 & 63u;   // row, byte-in-row
    const unsigned r1 = off1 >> 6, kb1 = off1 & 63u;
    const unsigned aL0 = (unsigned)(size_t)((unsigned short*)lds_a) + off0;
    const unsigned aL1 = (unsigned)(size_t)((unsigned short*)lds_a) + off1;
    const unsigned bL0 = (unsigned)(size_t)((unsigned short*)lds_b) + off0;
    const unsigned bL1 = (unsigned)(size_t)((unsigned short*)lds_b) + off1;

    auto stage = [&](int buf, int k0) {
        const unsigned bsel = (unsigned)buf * 8192u;
        const unsigned gk = (unsigned)k0 * 2u;
        async_cp_b128(aL0 + bsel, ((unsigned)(m0 + r0) * (unsigned)K) * 2u + gk + kb0, Abf);
        async_cp_b128(aL1 + bsel, ((unsigned)(m0 + r1) * (unsigned)K) * 2u + gk + kb1, Abf);
        async_cp_b128(bL0 + bsel, ((unsigned)(n0 + r0) * (unsigned)K) * 2u + gk + kb0, Bt);
        async_cp_b128(bL1 + bsel, ((unsigned)(n0 + r1) * (unsigned)K) * 2u + gk + kb1, Bt);
    };

    const v8f zero = {0.f,0.f,0.f,0.f,0.f,0.f,0.f,0.f};
    v8f acc[4][2];
#pragma unroll
    for (int i = 0; i < 4; ++i) { acc[i][0] = zero; acc[i][1] = zero; }

    const int ksteps = K >> 5;
    stage(0, 0);
    for (int ks = 0; ks < ksteps; ++ks) {
        const int cur = ks & 1;
        const int more = (ks + 1 < ksteps);
        if (more) stage(1 - cur, (ks + 1) << 5);
        wait_async(more);      // current tile's 4 copies complete (in-order)
        __syncthreads();

        const unsigned short* la = lds_a[cur];
        const unsigned short* lb = lds_b[cur];
        v16bf bfr[2];
#pragma unroll
        for (int j = 0; j < 2; ++j)
            bfr[j] = frag_ld(lb, wn + j * 16 + lrow, 32, 0, half);
#pragma unroll
        for (int i = 0; i < 4; ++i) {
            v16bf afr = frag_ld(la, wm + i * 16 + lrow, 32, 0, half);
            acc[i][0] = wmma_bf16(afr, bfr[0], acc[i][0]);
            acc[i][1] = wmma_bf16(afr, bfr[1], acc[i][1]);
        }
        __syncthreads();       // all waves done reading before next overwrite
    }

    // writeback: C/D layout -> lane<16: (m=r, n=lane); lane>=16: (m=8+r, n=lane-16)
#pragma unroll
    for (int i = 0; i < 4; ++i)
#pragma unroll
        for (int j = 0; j < 2; ++j)
#pragma unroll
            for (int r = 0; r < 8; ++r) {
                int m = m0 + wm + i * 16 + half * 8 + r;
                int n = n0 + wn + j * 16 + lrow;
                float v = acc[i][j][r];
                if constexpr (OUT_BF16)
                    ((unsigned short*)Cout)[(size_t)m * Nn + n] = f2bf(v);
                else
                    ((float*)Cout)[(size_t)m * Nn + n] = v + bias[n];
            }
}

// ---------------------------------------------------------------------------
// Flash-attention GQA: 64 query rows per block, 4 waves x 16-row strips,
// causal, online softmax in fp32, QK^T and PV via bf16 WMMA.
// Q/K/V bf16: Q[b,n,h,dh], K/V[b,n,kv,dh]. Ctx out bf16 [b,n,h,dh].
// ---------------------------------------------------------------------------
__global__ void __launch_bounds__(128)
fa_gqa(const unsigned short* __restrict__ Q,
       const unsigned short* __restrict__ Km,
       const unsigned short* __restrict__ V,
       unsigned short* __restrict__ Ctx)
{
    __shared__ __align__(16) unsigned short q_lds[64 * 64];   // [m][d]
    __shared__ __align__(16) unsigned short k_lds[64 * 64];   // [n][d]
    __shared__ __align__(16) unsigned short vt_lds[64 * 64];  // [d][n]
    __shared__ __align__(16) unsigned short p_lds[4 * 16 * 64];

    const int tid  = threadIdx.x;
    const int lane = tid & 31;
    const int w    = tid >> 5;
    const int half = lane >> 4;
    const int lrow = lane & 15;

    int wg = blockIdx.x;
    const int nblk = BQ_N / 64;
    const int qb = wg % nblk;  wg /= nblk;
    const int g  = wg % NG;    wg /= NG;
    const int kv = wg % NKV;   wg /= NKV;
    const int b  = wg;
    const int h  = kv * NG + g;
    const int m0 = qb * 64;

    // stage Q block (64x64)
#pragma unroll
    for (int i = 0; i < 8; ++i) {
        int idx = (tid + i * 128) * 4;
        int row = idx >> 6, d = idx & 63;
        *(v4us*)(q_lds + row * 64 + d) =
            *(const v4us*)(Q + (((size_t)(b * BQ_N + m0 + row) * NH + h) * DHD + d));
    }
    __syncthreads();

    v16bf qfr[2];
#pragma unroll
    for (int kk = 0; kk < 2; ++kk)
        qfr[kk] = frag_ld(q_lds, w * 16 + lrow, 64, kk * 32, half);

    const v8f zero = {0.f,0.f,0.f,0.f,0.f,0.f,0.f,0.f};
    float mrun[8], lrun[8];
    v8f accd[4];
#pragma unroll
    for (int r = 0; r < 8; ++r) { mrun[r] = -3.0e38f; lrun[r] = 0.f; }
#pragma unroll
    for (int dt = 0; dt < 4; ++dt) accd[dt] = zero;

    const float scale = 0.125f; // 1/sqrt(64)

    for (int kb = 0; kb <= qb; ++kb) {
        const int n0 = kb * 64;
        __syncthreads();
        // stage K [n][d] and V^T [d][n]
#pragma unroll
        for (int i = 0; i < 8; ++i) {
            int idx = (tid + i * 128) * 4;
            int row = idx >> 6, d = idx & 63;
            size_t ga = ((size_t)(b * BQ_N + n0 + row) * NKV + kv) * DHD + d;
            *(v4us*)(k_lds + row * 64 + d) = *(const v4us*)(Km + ga);
            v4us vv = *(const v4us*)(V + ga);
            vt_lds[(d + 0) * 64 + row] = vv.x;
            vt_lds[(d + 1) * 64 + row] = vv.y;
            vt_lds[(d + 2) * 64 + row] = vv.z;
            vt_lds[(d + 3) * 64 + row] = vv.w;
        }
        __syncthreads();

        // S strip (16 x 64) = Q K^T
        v8f s[4];
#pragma unroll
        for (int j = 0; j < 4; ++j) s[j] = zero;
#pragma unroll
        for (int kk = 0; kk < 2; ++kk)
#pragma unroll
            for (int j = 0; j < 4; ++j) {
                v16bf kf = frag_ld(k_lds, j * 16 + lrow, 64, kk * 32, half);
                s[j] = wmma_bf16(qfr[kk], kf, s[j]);
            }

        // scale + causal mask
        const int mg_base = m0 + w * 16 + half * 8;
#pragma unroll
        for (int j = 0; j < 4; ++j) {
            const int ng = n0 + j * 16 + lrow;
#pragma unroll
            for (int r = 0; r < 8; ++r) {
                float sv = s[j][r] * scale;
                if (ng > mg_base + r) sv = -3.0e38f;
                s[j][r] = sv;
            }
        }

        // online softmax row stats (rows live per half-wave; reduce over 16 lanes)
        float mnew[8], alpha[8];
#pragma unroll
        for (int r = 0; r < 8; ++r) {
            float mt = fmaxf(fmaxf(s[0][r], s[1][r]), fmaxf(s[2][r], s[3][r]));
#pragma unroll
            for (int sh = 1; sh < 16; sh <<= 1) mt = fmaxf(mt, __shfl_xor(mt, sh, 16));
            mnew[r]  = fmaxf(mrun[r], mt);
            alpha[r] = __expf(mrun[r] - mnew[r]);
            mrun[r]  = mnew[r];
        }
#pragma unroll
        for (int j = 0; j < 4; ++j)
#pragma unroll
            for (int r = 0; r < 8; ++r)
                s[j][r] = __expf(s[j][r] - mnew[r]);
#pragma unroll
        for (int r = 0; r < 8; ++r) {
            float t = s[0][r] + s[1][r] + s[2][r] + s[3][r];
#pragma unroll
            for (int sh = 1; sh < 16; sh <<= 1) t += __shfl_xor(t, sh, 16);
            lrun[r] = lrun[r] * alpha[r] + t;
        }
#pragma unroll
        for (int dt = 0; dt < 4; ++dt)
#pragma unroll
            for (int r = 0; r < 8; ++r)
                accd[dt][r] *= alpha[r];

        // P (C-layout) -> LDS -> A-layout fragments
        unsigned short* pw = p_lds + w * (16 * 64);
#pragma unroll
        for (int j = 0; j < 4; ++j)
#pragma unroll
            for (int r = 0; r < 8; ++r)
                pw[(half * 8 + r) * 64 + j * 16 + lrow] = f2bf(s[j][r]);
        __syncthreads();

        // O += P V
#pragma unroll
        for (int kk = 0; kk < 2; ++kk) {
            v16bf pf = frag_ld(pw, lrow, 64, kk * 32, half);
#pragma unroll
            for (int dt = 0; dt < 4; ++dt) {
                v16bf vf = frag_ld(vt_lds, dt * 16 + lrow, 64, kk * 32, half);
                accd[dt] = wmma_bf16(pf, vf, accd[dt]);
            }
        }
    }

    // normalize and store ctx (bf16)
#pragma unroll
    for (int dt = 0; dt < 4; ++dt)
#pragma unroll
        for (int r = 0; r < 8; ++r) {
            int m = m0 + w * 16 + half * 8 + r;
            int d = dt * 16 + lrow;
            Ctx[((size_t)(b * BQ_N + m) * NH + h) * DHD + d] =
                f2bf(accd[dt][r] / lrun[r]);
        }
}

// ---------------------------------------------------------------------------
extern "C" void kernel_launch(void* const* d_in, const int* in_sizes, int n_in,
                              void* d_out, int out_size, void* d_ws, size_t ws_size,
                              hipStream_t stream)
{
    (void)in_sizes; (void)n_in; (void)out_size; (void)ws_size;
    const float* x  = (const float*)d_in[0];
    const float* Wq = (const float*)d_in[1];
    const float* Wk = (const float*)d_in[2];
    const float* Wv = (const float*)d_in[3];
    const float* Wo = (const float*)d_in[4];
    const float* bo = (const float*)d_in[5];
    float* out = (float*)d_out;

    const int M   = BQ_B * BQ_N;     // 4096
    const int NKD = NKV * DHD;       // 512

    unsigned short* xb  = (unsigned short*)d_ws;         // x bf16          16 MB
    unsigned short* Wqt = xb  + (size_t)M * D_IN;        // Wq^T [2048][2048] 8 MB
    unsigned short* Wkt = Wqt + (size_t)D_OUT * D_IN;    // Wk^T [512][2048]  2 MB
    unsigned short* Wvt = Wkt + (size_t)NKD * D_IN;      // Wv^T [512][2048]  2 MB
    unsigned short* Wot = Wvt + (size_t)NKD * D_IN;      // Wo^T [2048][2048] 8 MB
    unsigned short* qb  = Wot + (size_t)D_IN * D_OUT;    // Q bf16           16 MB
    unsigned short* kb  = qb  + (size_t)M * D_OUT;       // K bf16            4 MB
    unsigned short* vb  = kb  + (size_t)M * NKD;         // V bf16            4 MB
    unsigned short* cb  = vb  + (size_t)M * NKD;         // ctx bf16         16 MB

    dim3 blk(256);
    // pre-pass: x -> bf16; weights -> bf16 transposed [n][k]
    cvt_f32_bf16<<<dim3((M * D_IN) / 1024), blk, 0, stream>>>(x, xb);
    transpose_f32_bf16<<<dim3(D_OUT / 64, D_IN / 64), blk, 0, stream>>>(Wq, Wqt, D_IN, D_OUT);
    transpose_f32_bf16<<<dim3(NKD / 64,  D_IN / 64), blk, 0, stream>>>(Wk, Wkt, D_IN, NKD);
    transpose_f32_bf16<<<dim3(NKD / 64,  D_IN / 64), blk, 0, stream>>>(Wv, Wvt, D_IN, NKD);
    transpose_f32_bf16<<<dim3(D_IN / 64, D_OUT / 64), blk, 0, stream>>>(Wo, Wot, D_OUT, D_IN);

    // projections (bf16 out)
    gemm_wmma_async<true><<<dim3(D_OUT / 128, M / 128), blk, 0, stream>>>(
        xb, Wqt, nullptr, qb, M, D_OUT, D_IN);
    gemm_wmma_async<true><<<dim3(NKD / 128, M / 128), blk, 0, stream>>>(
        xb, Wkt, nullptr, kb, M, NKD, D_IN);
    gemm_wmma_async<true><<<dim3(NKD / 128, M / 128), blk, 0, stream>>>(
        xb, Wvt, nullptr, vb, M, NKD, D_IN);

    // attention
    fa_gqa<<<dim3(BQ_B * NKV * NG * (BQ_N / 64)), dim3(128), 0, stream>>>(
        qb, kb, vb, cb);

    // output projection (fp32 + bias)
    gemm_wmma_async<false><<<dim3(D_IN / 128, M / 128), blk, 0, stream>>>(
        cb, Wot, bo, out, M, D_IN, D_OUT);
}
